// UnquantizedSparseMoELayer_82076825027369
// MI455X (gfx1250) — compile-verified, matrix operand order checked by reading, not run on hardware
//
#include <hip/hip_runtime.h>
#include <hip/hip_bf16.h>

typedef __attribute__((ext_vector_type(16))) __bf16 v16bf;
typedef __attribute__((ext_vector_type(8)))  float  v8f;

#define TOKENS  2048
#define HIDDEN  1024
#define NEXP    8
#define INTER   2816
#define BLOCK_M 32
#define CHUNK   256   // intermediate channels per fused chunk (8 waves x 32)
#define ASTR    272   // bf16 stride for act buffer: 544B rows, 32B-aligned

// ---- helpers -------------------------------------------------------------

__device__ __forceinline__ v16bf ld16f_bf(const float* __restrict__ p) {
  // load 16 consecutive fp32 (64B, 16B-aligned) and convert to bf16 vector
  const float4* q = (const float4*)p;
  float4 f0 = q[0], f1 = q[1], f2 = q[2], f3 = q[3];
  v16bf r;
  r[0]=(__bf16)f0.x; r[1]=(__bf16)f0.y; r[2]=(__bf16)f0.z; r[3]=(__bf16)f0.w;
  r[4]=(__bf16)f1.x; r[5]=(__bf16)f1.y; r[6]=(__bf16)f1.z; r[7]=(__bf16)f1.w;
  r[8]=(__bf16)f2.x; r[9]=(__bf16)f2.y; r[10]=(__bf16)f2.z; r[11]=(__bf16)f2.w;
  r[12]=(__bf16)f3.x; r[13]=(__bf16)f3.y; r[14]=(__bf16)f3.z; r[15]=(__bf16)f3.w;
  return r;
}

__device__ __forceinline__ v8f wmma_bf16(v16bf a, v16bf b, v8f c) {
  // D = A(16x32) * B(32x16) + C, fp32 accumulate
  return __builtin_amdgcn_wmma_f32_16x16x32_bf16(
      /*neg_a=*/false, a, /*neg_b=*/false, b,
      /*c_mod=*/(short)0, c, /*reuse_a=*/false, /*reuse_b=*/false);
}

// ---- kernel 0: zero per-expert counters ---------------------------------

__global__ void moe_init(int* __restrict__ cnt) {
  if (threadIdx.x < NEXP) cnt[threadIdx.x] = 0;
}

// ---- kernel 1: routing (softmax top-2, renormalize, build expert lists) --

__global__ void moe_route(const float* __restrict__ gating,
                          float* __restrict__ tw,
                          int* __restrict__ cnt,
                          int* __restrict__ list) {
  int t = blockIdx.x * blockDim.x + threadIdx.x;
  if (t >= TOKENS) return;
  float l[NEXP];
#pragma unroll
  for (int i = 0; i < NEXP; ++i) l[i] = gating[t * NEXP + i];
  int i0 = 0; float b0 = l[0];
#pragma unroll
  for (int i = 1; i < NEXP; ++i) if (l[i] > b0) { b0 = l[i]; i0 = i; }
  int i1 = -1; float b1 = -3.0e38f;
#pragma unroll
  for (int i = 0; i < NEXP; ++i) if (i != i0 && l[i] > b1) { b1 = l[i]; i1 = i; }
  // renormalized top-2 softmax == softmax over the two logits
  float w0 = 1.0f / (1.0f + __expf(b1 - b0));
  tw[2 * t + 0] = w0;
  tw[2 * t + 1] = 1.0f - w0;
  int p0 = atomicAdd(&cnt[i0], 1); list[i0 * TOKENS + p0] = 2 * t + 0;
  int p1 = atomicAdd(&cnt[i1], 1); list[i1 * TOKENS + p1] = 2 * t + 1;
}

// ---- kernel 1b: pre-convert activations to bf16 (A-operand, zero-cvt) ----

__global__ void moe_cvt_x(const float* __restrict__ x, __bf16* __restrict__ xb) {
  int i = blockIdx.x * 256 + threadIdx.x;   // TOKENS*HIDDEN threads
  xb[i] = (__bf16)x[i];
}

// ---- kernel 2: fused expert FFN (w1 -> silu*up -> w2), bf16 WMMA ---------

__global__ __launch_bounds__(256, 1)
void moe_ffn(const __bf16* __restrict__ xb,
             const float* __restrict__ w1,
             const float* __restrict__ w2,
             const float* __restrict__ tw,
             const int*   __restrict__ cnt,
             const int*   __restrict__ list,
             float*       __restrict__ slot) {
  __shared__ __bf16 acts[BLOCK_M * ASTR];  // silu(gate)*up, one 256-ch chunk
  __shared__ int    ent_s[BLOCK_M];
  __shared__ float  w_s[BLOCK_M];

  const int e     = blockIdx.y;
  const int n     = cnt[e];
  const int tile0 = blockIdx.x * BLOCK_M;
  if (tile0 >= n) return;

  const int tid  = threadIdx.x;
  const int lane = tid & 31;
  const int wave = tid >> 5;
  const int ln   = lane & 15;         // N-column / M-row within tile
  const int kh   = (lane >> 4) << 4;  // 0 or 16: K-half owned by this lane
  const int hb   = (lane >> 4) << 3;  // 0 or 8: C-row offset per VGPR layout

  if (tid < BLOCK_M) {
    int r = tile0 + tid;
    if (r < n) { int ent = list[e * TOKENS + r]; ent_s[tid] = ent; w_s[tid] = tw[ent]; }
    else       { ent_s[tid] = -1; w_s[tid] = 0.0f; }
  }
  __syncthreads();

  // A-operand row pointers (bf16 x rows of this lane's two M-tiles)
  const int e0 = ent_s[ln], e1 = ent_s[16 + ln];
  const __bf16* xb0 = xb + (size_t)((e0 >= 0) ? (e0 >> 1) : 0) * HIDDEN + kh;
  const __bf16* xb1 = xb + (size_t)((e1 >= 0) ? (e1 >> 1) : 0) * HIDDEN + kh;

  // FFN1: this wave owns gate+up for channels [32*wave, 32*wave+32) of chunk
  const size_t w1rowc = (size_t)e * 2 * INTER + 32 * (size_t)wave + ln;
  // FFN2: wave owns output channels h0..h0+127
  const int h0 = wave * 128;
  const float* w2b = w2 + ((size_t)e * HIDDEN + h0 + ln) * INTER + kh;

  // persistent FFN2 accumulators: 2 M-tiles x 8 N-tiles (128 VGPRs fp32)
  v8f acc[2][8];
#pragma unroll
  for (int mt = 0; mt < 2; ++mt)
#pragma unroll
    for (int nt = 0; nt < 8; ++nt) { v8f z = {}; acc[mt][nt] = z; }

  for (int c = 0; c < INTER / CHUNK; ++c) {   // 11 chunks
    const int i0 = c * CHUNK;

    // --- FFN1: gate & up, 2M x 2N tiles each, K=1024 ----------------------
    v8f g[2][2], u[2][2];
#pragma unroll
    for (int mt = 0; mt < 2; ++mt)
#pragma unroll
      for (int nt = 0; nt < 2; ++nt) { v8f z = {}; g[mt][nt] = z; u[mt][nt] = z; }

    const float* gbp = w1 + (w1rowc + (size_t)i0) * HIDDEN + kh;
    const float* ubp = gbp + (size_t)INTER * HIDDEN;
#pragma unroll 2
    for (int kk = 0; kk < HIDDEN / 32; ++kk) {
      v16bf a0  = *(const v16bf*)(xb0 + kk * 32);
      v16bf a1  = *(const v16bf*)(xb1 + kk * 32);
      v16bf bg0 = ld16f_bf(gbp + kk * 32);
      v16bf bg1 = ld16f_bf(gbp + (size_t)16 * HIDDEN + kk * 32);
      v16bf bu0 = ld16f_bf(ubp + kk * 32);
      v16bf bu1 = ld16f_bf(ubp + (size_t)16 * HIDDEN + kk * 32);
      g[0][0] = wmma_bf16(a0, bg0, g[0][0]);
      g[0][1] = wmma_bf16(a0, bg1, g[0][1]);
      g[1][0] = wmma_bf16(a1, bg0, g[1][0]);
      g[1][1] = wmma_bf16(a1, bg1, g[1][1]);
      u[0][0] = wmma_bf16(a0, bu0, u[0][0]);
      u[0][1] = wmma_bf16(a0, bu1, u[0][1]);
      u[1][0] = wmma_bf16(a1, bu0, u[1][0]);
      u[1][1] = wmma_bf16(a1, bu1, u[1][1]);
    }

    // --- act = silu(gate) * up, in-register, publish bf16 -----------------
#pragma unroll
    for (int mt = 0; mt < 2; ++mt)
#pragma unroll
      for (int nt = 0; nt < 2; ++nt)
#pragma unroll
        for (int r = 0; r < 8; ++r) {
          int row = 16 * mt + r + hb;
          int col = 32 * wave + 16 * nt + ln;
          float gg = g[mt][nt][r], uu = u[mt][nt][r];
          acts[row * ASTR + col] = (__bf16)(gg / (1.0f + __expf(-gg)) * uu);
        }
    __syncthreads();

    // --- FFN2: acc += act(32x256) * w2[e][h0..h0+127][i0..i0+255]^T -------
#pragma unroll
    for (int kk = 0; kk < CHUNK / 32; ++kk) {   // 8 K-steps
      v16bf a0 = *(const v16bf*)&acts[ln * ASTR + kk * 32 + kh];
      v16bf a1 = *(const v16bf*)&acts[(16 + ln) * ASTR + kk * 32 + kh];
#pragma unroll
      for (int nt = 0; nt < 8; ++nt) {
        v16bf b = ld16f_bf(w2b + (size_t)(nt * 16) * INTER + i0 + kk * 32);
        acc[0][nt] = wmma_bf16(a0, b, acc[0][nt]);
        acc[1][nt] = wmma_bf16(a1, b, acc[1][nt]);
      }
    }
    __syncthreads();   // protect acts against next chunk's writers
  }

  // --- weighted scatter into per-(token,slot) buffer (unique slots) -------
#pragma unroll
  for (int mt = 0; mt < 2; ++mt)
#pragma unroll
    for (int nt = 0; nt < 8; ++nt)
#pragma unroll
      for (int r = 0; r < 8; ++r) {
        int row = mt * 16 + r + hb;
        int ent = ent_s[row];
        if (ent >= 0)
          slot[(size_t)ent * HIDDEN + h0 + nt * 16 + ln] = acc[mt][nt][r] * w_s[row];
      }
}

// ---- kernel 3: combine the two expert slots per token --------------------

__global__ void moe_combine(const float* __restrict__ slot, float* __restrict__ out) {
  int i = blockIdx.x * 256 + threadIdx.x;
  if (i >= TOKENS * HIDDEN) return;
  size_t t = (size_t)(i >> 10);  // HIDDEN = 1024
  out[i] = slot[(size_t)i + t * HIDDEN] + slot[(size_t)i + t * HIDDEN + HIDDEN];
}

// ---- launch --------------------------------------------------------------

extern "C" void kernel_launch(void* const* d_in, const int* in_sizes, int n_in,
                              void* d_out, int out_size, void* d_ws, size_t ws_size,
                              hipStream_t stream) {
  const float* x      = (const float*)d_in[0];
  const float* gating = (const float*)d_in[1];
  const float* w1     = (const float*)d_in[2];
  const float* w2     = (const float*)d_in[3];
  float* out = (float*)d_out;

  char* ws = (char*)d_ws;
  int*    cnt  = (int*)ws;                                   // 8 ints
  int*    list = (int*)(ws + 256);                           // 8*2048 ints
  float*  tw   = (float*)(ws + 256 + NEXP * TOKENS * 4);     // 4096 floats
  __bf16* xbf  = (__bf16*)(ws + 98304);                      // 2048*1024 bf16 (4 MB)
  float*  slot = (float*)(ws + 98304 + (size_t)TOKENS * HIDDEN * 2); // 16.8 MB

  moe_init<<<1, 64, 0, stream>>>(cnt);
  moe_route<<<TOKENS / 256, 256, 0, stream>>>(gating, tw, cnt, list);
  moe_cvt_x<<<(TOKENS * HIDDEN) / 256, 256, 0, stream>>>(x, xbf);
  moe_ffn<<<dim3(TOKENS / BLOCK_M, NEXP), 256, 0, stream>>>(xbf, w1, w2, tw, cnt, list, slot);
  moe_combine<<<(TOKENS * HIDDEN) / 256, 256, 0, stream>>>(slot, out);
}
